// SelfAttentionLayer_6863357739620
// MI455X (gfx1250) — matrix operand order repaired and checked by reading, not compiled, not run on hardware
//
#include <hip/hip_runtime.h>
#include <hip/hip_bf16.h>
#include <math.h>

// ---------- constants ----------
#define HIDDEN 1024
#define NH     16
#define HD     64
#define TSEQ   2048
#define LOG2E  1.4426950408889634f
#define LOG2_10000 13.287712379549449f

// ---------- WMMA types ----------
typedef __bf16 bf16_t;
typedef bf16_t v16bf __attribute__((ext_vector_type(16)));
typedef float  v8f   __attribute__((ext_vector_type(8)));

union FragBF {
    v16bf v;
    uint4 q[2];
    unsigned short u[16];
};

__device__ __forceinline__ unsigned short f2bf(float f) {
    unsigned int u = __float_as_uint(f);
    unsigned int r = u + 0x7FFFu + ((u >> 16) & 1u);
    return (unsigned short)(r >> 16);
}
__device__ __forceinline__ float bf2f(unsigned short h) {
    return __uint_as_float(((unsigned int)h) << 16);
}
// pack two fp32 -> two bf16 in one dword (low = a, high = b)
__device__ __forceinline__ unsigned int pk2bf(float a, float b) {
#if __has_builtin(__builtin_amdgcn_cvt_pk_bf16_f32)
    typedef __bf16 v2bf __attribute__((ext_vector_type(2)));
    v2bf p = __builtin_amdgcn_cvt_pk_bf16_f32(a, b);
    return __builtin_bit_cast(unsigned int, p);
#else
    return (unsigned int)f2bf(a) | ((unsigned int)f2bf(b) << 16);
#endif
}

__device__ __forceinline__ v8f wmma_bf16(const v16bf& a, const v16bf& b, const v8f& c) {
    return __builtin_amdgcn_wmma_f32_16x16x32_bf16(
        /*neg_a=*/false, a, /*neg_b=*/false, b,
        /*c_mod=*/(short)0, c, /*reuse_a=*/false, /*reuse_b=*/false);
}

// =====================================================================
// GEMM: C[M,N] = A[M,K] * B[K,N] + bias, bf16 WMMA, f32 accumulate.
// MODE 0: A = fp32 (x), epilogue scatters bf16 to head-major Q/K/V ws.
// MODE 1: A = bf16 (attn ws), epilogue writes fp32 to d_out.
// Block tile: 128(M) x 128(N), K-step 32, 8 waves (4Mx2N),
// each wave 32x64 = 2x4 WMMA tiles -> 8 WMMAs per K-step.
// =====================================================================
template <int MODE>
__global__ __launch_bounds__(256)
void gemm_wmma(const float* __restrict__ Af32,
               const unsigned short* __restrict__ Abf16,
               const float* __restrict__ Bmat,
               const float* __restrict__ bias,
               unsigned short* __restrict__ q_out,
               unsigned short* __restrict__ k_out,
               unsigned short* __restrict__ v_out,
               float* __restrict__ out_f32,
               int M, int N, int K) {
    __shared__ __align__(16) unsigned short sA[128 * 40]; // [row][k], pad 8
    __shared__ __align__(16) unsigned short sB[128 * 40]; // [col][k] (B^T), pad 8

    const int tid  = threadIdx.x;
    const int lane = tid & 31;
    const int w    = tid >> 5;     // 0..7
    const int wm   = w & 3;        // M strip (32 rows)
    const int wn   = w >> 2;       // N strip (64 cols)

    const int m0 = blockIdx.y * 128;
    const int n0 = blockIdx.x * 128;

    v8f acc[2][4];
    const v8f vzero = {0.f, 0.f, 0.f, 0.f, 0.f, 0.f, 0.f, 0.f};
#pragma unroll
    for (int mi = 0; mi < 2; ++mi)
#pragma unroll
        for (int ni = 0; ni < 4; ++ni) acc[mi][ni] = vzero;

    // A staging: thread covers 4 consecutive k of one row, 4 row-passes
    const int tr = tid >> 3;          // 0..31
    const int tk = (tid & 7) * 4;     // 0..28
    // B staging: float4 along N, transpose into sB, 4 k-passes
    const int bn4 = (tid & 31) * 4;   // 0..124
    const int bkk = tid >> 5;         // 0..7

    for (int k0 = 0; k0 < K; k0 += 32) {
        // ---- stage A tile (128x32) as bf16 ----
#pragma unroll
        for (int p = 0; p < 4; ++p) {
            int row  = p * 32 + tr;
            size_t g = (size_t)(m0 + row) * K + (k0 + tk);
            if (MODE == 0) {
                float4 f = *(const float4*)(Af32 + g);
                *(unsigned int*)&sA[row * 40 + tk + 0] = pk2bf(f.x, f.y);
                *(unsigned int*)&sA[row * 40 + tk + 2] = pk2bf(f.z, f.w);
            } else {
                uint2 u = *(const uint2*)(Abf16 + g);
                *(uint2*)&sA[row * 40 + tk] = u;
            }
        }
        // ---- stage B tile (32x128) transposed as bf16 ----
#pragma unroll
        for (int p = 0; p < 4; ++p) {
            int kk   = p * 8 + bkk;
            float4 f = *(const float4*)(Bmat + (size_t)(k0 + kk) * N + (n0 + bn4));
            sB[(bn4 + 0) * 40 + kk] = f2bf(f.x);
            sB[(bn4 + 1) * 40 + kk] = f2bf(f.y);
            sB[(bn4 + 2) * 40 + kk] = f2bf(f.z);
            sB[(bn4 + 3) * 40 + kk] = f2bf(f.w);
        }
        // ---- prefetch next K tiles ----
        if (k0 + 32 < K) {
            if (MODE == 0)
                __builtin_prefetch(Af32 + (size_t)(m0 + tr) * K + (k0 + 32 + tk), 0, 1);
            else
                __builtin_prefetch(Abf16 + (size_t)(m0 + tr) * K + (k0 + 32 + tk), 0, 1);
            __builtin_prefetch(Bmat + (size_t)(k0 + 32 + bkk) * N + (n0 + bn4), 0, 1);
        }
        __syncthreads();

        // ---- fragments + 8 WMMAs ----
        FragBF afrag[2], bfrag[4];
#pragma unroll
        for (int mi = 0; mi < 2; ++mi) {
            int row = wm * 32 + mi * 16 + (lane & 15);
            int kg  = (lane >> 4) * 8;
            afrag[mi].q[0] = *(const uint4*)&sA[row * 40 + kg];
            afrag[mi].q[1] = *(const uint4*)&sA[row * 40 + kg + 16];
        }
#pragma unroll
        for (int ni = 0; ni < 4; ++ni) {
            int col = wn * 64 + ni * 16 + (lane & 15);
            int kg  = (lane >> 4) * 8;
            bfrag[ni].q[0] = *(const uint4*)&sB[col * 40 + kg];
            bfrag[ni].q[1] = *(const uint4*)&sB[col * 40 + kg + 16];
        }
#pragma unroll
        for (int mi = 0; mi < 2; ++mi)
#pragma unroll
            for (int ni = 0; ni < 4; ++ni)
                acc[mi][ni] = wmma_bf16(afrag[mi].v, bfrag[ni].v, acc[mi][ni]);

        __syncthreads();
    }

    // ---- epilogue ----
#pragma unroll
    for (int mi = 0; mi < 2; ++mi) {
#pragma unroll
        for (int ni = 0; ni < 4; ++ni) {
#pragma unroll
            for (int v = 0; v < 8; ++v) {
                int row = m0 + wm * 32 + mi * 16 + v + 8 * (lane >> 4);
                int col = n0 + wn * 64 + ni * 16 + (lane & 15);
                float val = acc[mi][ni][v] + bias[col];
                if (MODE == 0) {
                    int sect = col >> 10;          // 0=q 1=k 2=v
                    int h    = (col >> 6) & 15;
                    int d    = col & 63;
                    int b    = row >> 11;          // /TSEQ
                    int t    = row & 2047;
                    unsigned short* dst = (sect == 0) ? q_out : ((sect == 1) ? k_out : v_out);
                    dst[(((size_t)b * NH + h) * TSEQ + t) * HD + d] = f2bf(val);
                } else {
                    out_f32[(size_t)row * N + col] = val;
                }
            }
        }
    }
}

// =====================================================================
// RoPE in place on head-major bf16 Q and K: pair (d, d+32) within head.
// =====================================================================
__global__ void rope_kernel(unsigned short* __restrict__ Qh,
                            unsigned short* __restrict__ Kh, int total) {
    int i = blockIdx.x * blockDim.x + threadIdx.x;
    if (i >= 2 * total) return;
    unsigned short* arr = (i < total) ? Qh : Kh;
    int j = (i < total) ? i : (i - total);
    int pair = j & 31;                 // 0..31
    int t    = (j >> 5) & (TSEQ - 1);  // position
    size_t rowbase = (size_t)(j >> 5) * HD;
    float freq = exp2f(-(float)(2 * pair) * (LOG2_10000 / (float)HD));
    float ang  = (float)t * freq;
    float s, c;
    sincosf(ang, &s, &c);
    float x1 = bf2f(arr[rowbase + pair]);
    float x2 = bf2f(arr[rowbase + 32 + pair]);
    arr[rowbase + pair]      = f2bf(x1 * c - x2 * s);
    arr[rowbase + 32 + pair] = f2bf(x1 * s + x2 * c);
}

// =====================================================================
// Flash attention (causal, online softmax). Block = (b,h) x 128 q rows,
// 8 waves each own 16 rows. Q/K^T frags direct from global (b128),
// V transposed through LDS, P bounced through wave-private LDS.
// =====================================================================
__global__ __launch_bounds__(256)
void flash_attn(const unsigned short* __restrict__ Qh,
                const unsigned short* __restrict__ Kh,
                const unsigned short* __restrict__ Vh,
                unsigned short* __restrict__ Oh) {
    __shared__ __align__(16) unsigned short sV[64 * 40];      // [dim][key] V^T
    __shared__ __align__(16) unsigned short sP[8 * 16 * 40];  // per-wave P

    const int tid  = threadIdx.x;
    const int lane = tid & 31;
    const int w    = tid >> 5;
    const int q0   = blockIdx.x * 128;
    const int bh   = blockIdx.y;
    const int b    = bh >> 4;
    const int h    = bh & 15;
    const size_t base = (size_t)bh * TSEQ * HD;

    const int qrow0 = q0 + w * 16;

    // resident Q fragments (2 K-steps over head dim 64)
    FragBF qf[2];
    {
        const unsigned short* qptr = Qh + base + (size_t)(qrow0 + (lane & 15)) * HD;
#pragma unroll
        for (int s = 0; s < 2; ++s) {
            int kb = s * 32 + (lane >> 4) * 8;
            qf[s].q[0] = *(const uint4*)(qptr + kb);
            qf[s].q[1] = *(const uint4*)(qptr + kb + 16);
        }
    }

    const v8f vzero = {0.f, 0.f, 0.f, 0.f, 0.f, 0.f, 0.f, 0.f};
    v8f acc[4];
#pragma unroll
    for (int dt = 0; dt < 4; ++dt) acc[dt] = vzero;
    float mrow[8], lrow[8];
#pragma unroll
    for (int v = 0; v < 8; ++v) { mrow[v] = -INFINITY; lrow[v] = 0.f; }

    const float cscale = 0.125f * LOG2E; // 1/sqrt(64) folded with log2(e)
    const int nkt = (q0 + 128) / 32;     // causal: keys up to q0+127

    for (int kt = 0; kt < nkt; ++kt) {
        const int k0 = kt * 32;

        // ---- stage V^T tile (32 keys x 64 dims -> [dim][key]) ----
        __syncthreads();
        {
            int key = tid >> 3;          // 0..31
            int d8  = (tid & 7) * 8;     // 0..56
            const unsigned short* vp = Vh + base + (size_t)(k0 + key) * HD + d8;
            uint4 u = *(const uint4*)vp;
            const unsigned short* us = (const unsigned short*)&u;
#pragma unroll
            for (int j = 0; j < 8; ++j) sV[(d8 + j) * 40 + key] = us[j];
            // prefetch next K/V tiles
            if (kt + 1 < nkt) {
                __builtin_prefetch(Vh + base + (size_t)(k0 + 32 + key) * HD + d8, 0, 1);
                __builtin_prefetch(Kh + base + (size_t)(k0 + 32 + key) * HD + d8, 0, 1);
            }
        }
        __syncthreads();

        const bool active = (k0 <= qrow0 + 15);  // wave-uniform
        if (active) {
            // ---- S = Q K^T (2 N-tiles x 2 K-steps) ----
            v8f s[2] = {vzero, vzero};
#pragma unroll
            for (int nt = 0; nt < 2; ++nt) {
#pragma unroll
                for (int sk = 0; sk < 2; ++sk) {
                    FragBF kf;
                    int key = k0 + nt * 16 + (lane & 15);
                    int kb  = sk * 32 + (lane >> 4) * 8;
                    const unsigned short* kp = Kh + base + (size_t)key * HD + kb;
                    kf.q[0] = *(const uint4*)kp;
                    kf.q[1] = *(const uint4*)(kp + 16);
                    s[nt] = wmma_bf16(qf[sk].v, kf.v, s[nt]);
                }
            }

            // ---- scale + causal mask + row max ----
            float mnew[8];
#pragma unroll
            for (int v = 0; v < 8; ++v) {
                int row = qrow0 + v + 8 * (lane >> 4);
                float mx = mrow[v];
#pragma unroll
                for (int nt = 0; nt < 2; ++nt) {
                    int col = k0 + nt * 16 + (lane & 15);
                    float val = s[nt][v] * cscale;
                    val = (col <= row) ? val : -INFINITY;
                    s[nt][v] = val;
                    mx = fmaxf(mx, val);
                }
#pragma unroll
                for (int off = 1; off < 16; off <<= 1)
                    mx = fmaxf(mx, __shfl_xor(mx, off, 32));
                mnew[v] = mx;
            }

            // ---- rescale + exponentiate + row sum ----
            float alpha[8];
#pragma unroll
            for (int v = 0; v < 8; ++v) {
                alpha[v] = exp2f(mrow[v] - mnew[v]);
                mrow[v]  = mnew[v];
            }
#pragma unroll
            for (int v = 0; v < 8; ++v) {
                float rs = 0.f;
#pragma unroll
                for (int nt = 0; nt < 2; ++nt) {
                    float p = exp2f(s[nt][v] - mnew[v]);
                    s[nt][v] = p;
                    rs += p;
                }
#pragma unroll
                for (int off = 1; off < 16; off <<= 1)
                    rs += __shfl_xor(rs, off, 32);
                lrow[v] = lrow[v] * alpha[v] + rs;
            }
#pragma unroll
            for (int dt = 0; dt < 4; ++dt)
#pragma unroll
                for (int v = 0; v < 8; ++v) acc[dt][v] *= alpha[v];

            // ---- bounce P through wave-private LDS to A-layout ----
            const int pbase = w * 640; // 16*40
#pragma unroll
            for (int nt = 0; nt < 2; ++nt)
#pragma unroll
                for (int v = 0; v < 8; ++v) {
                    int rr = v + 8 * (lane >> 4);
                    sP[pbase + rr * 40 + nt * 16 + (lane & 15)] = f2bf(s[nt][v]);
                }
            FragBF pf;
            {
                int pr = lane & 15;
                int pg = (lane >> 4) * 8;
                pf.q[0] = *(const uint4*)&sP[pbase + pr * 40 + pg];
                pf.q[1] = *(const uint4*)&sP[pbase + pr * 40 + pg + 16];
            }

            // ---- O += P V (K=32 keys, 4 dim tiles) ----
#pragma unroll
            for (int dt = 0; dt < 4; ++dt) {
                FragBF vf;
                int dcol = dt * 16 + (lane & 15);
                int kg   = (lane >> 4) * 8;
                vf.q[0] = *(const uint4*)&sV[dcol * 40 + kg];
                vf.q[1] = *(const uint4*)&sV[dcol * 40 + kg + 16];
                acc[dt] = wmma_bf16(pf.v, vf.v, acc[dt]);
            }
        }
    }

    // ---- normalize + store to [b][t][h][d] bf16 ws ----
    float linv[8];
#pragma unroll
    for (int v = 0; v < 8; ++v) linv[v] = 1.0f / lrow[v];
#pragma unroll
    for (int dt = 0; dt < 4; ++dt) {
#pragma unroll
        for (int v = 0; v < 8; ++v) {
            int row = qrow0 + v + 8 * (lane >> 4);
            int d   = dt * 16 + (lane & 15);
            float o = acc[dt][v] * linv[v];
            Oh[(((size_t)b * TSEQ + row) * NH + h) * HD + d] = f2bf(o);
        }
    }
}

// =====================================================================
__global__ void copy_f32(const float* __restrict__ src, float* __restrict__ dst, int n) {
    int i = blockIdx.x * blockDim.x + threadIdx.x;
    if (i < n) dst[i] = src[i];
}

// =====================================================================
extern "C" void kernel_launch(void* const* d_in, const int* in_sizes, int n_in,
                              void* d_out, int out_size, void* d_ws, size_t ws_size,
                              hipStream_t stream) {
    const float* x     = (const float*)d_in[0];
    const float* rnn   = (const float*)d_in[1];
    const float* Wqkv  = (const float*)d_in[2];
    const float* bqkv  = (const float*)d_in[3];
    const float* Wproj = (const float*)d_in[4];
    const float* bproj = (const float*)d_in[5];

    const int B = in_sizes[0] / (TSEQ * HIDDEN);
    const int M = B * TSEQ;

    const size_t sz = (size_t)B * NH * TSEQ * HD; // elements per head-major tensor
    unsigned short* qh   = (unsigned short*)d_ws;
    unsigned short* kh   = qh + sz;
    unsigned short* vh   = kh + sz;
    unsigned short* attn = vh + sz;

    float* out = (float*)d_out;

    // 1) QKV projection -> head-major bf16 q/k/v
    gemm_wmma<0><<<dim3(3 * HIDDEN / 128, M / 128), 256, 0, stream>>>(
        x, nullptr, Wqkv, bqkv, qh, kh, vh, nullptr, M, 3 * HIDDEN, HIDDEN);

    // 2) RoPE on q and k
    {
        int total = B * NH * TSEQ * 32;
        int n = 2 * total;
        rope_kernel<<<(n + 255) / 256, 256, 0, stream>>>(qh, kh, total);
    }

    // 3) causal flash attention -> attn ws (bf16, [b,t,h,d] == [M,1024])
    flash_attn<<<dim3(TSEQ / 128, B * NH), 256, 0, stream>>>(qh, kh, vh, attn);

    // 4) output projection -> d_out (fp32)
    gemm_wmma<1><<<dim3(HIDDEN / 128, M / 128), 256, 0, stream>>>(
        nullptr, attn, Wproj, bproj, nullptr, nullptr, nullptr, out, M, HIDDEN, HIDDEN);

    // 5) rnn_state passthrough
    {
        int n = B * HIDDEN;
        copy_f32<<<(n + 255) / 256, 256, 0, stream>>>(rnn, out + (size_t)M * HIDDEN, n);
    }
}
